// GS_Sentence_74148315398665
// MI455X (gfx1250) — matrix operand order repaired and checked by reading, not compiled, not run on hardware
//
#include <hip/hip_runtime.h>

// Problem constants (match reference)
#define D_  300
#define H_  256
#define B_  64
#define S_  512
#define NB_ 5
#define DP_ 320          // D padded to multiple of 32 (zero-filled)
#define KP_ 576          // H + DP : fused [h | x] K dimension
#define G4_ 1024         // 4*H gate width
#define H2_ 512          // 2*H (concat fwd/bwd)
#define NBD_ 10          // NB branches * 2 directions

typedef _Float16 h16;
typedef __attribute__((ext_vector_type(16))) _Float16 v16h;
typedef __attribute__((ext_vector_type(8)))  _Float16 v8h;
typedef __attribute__((ext_vector_type(8)))  float    v8f;

// Global (addrspace 1) pointer types: guarantee global_load/global_store
// (FLAT ops would tie up DScnt + the LDS datapath; see ISA 10.1.1).
typedef __attribute__((address_space(1))) const h16  gc_h16;
typedef __attribute__((address_space(1))) const v16h gc_v16h;
typedef __attribute__((address_space(1))) const v8h  gc_v8h;
typedef __attribute__((address_space(1)))       h16  g_h16;

union AF { v16h v16; v8h v8[2]; };

#if __has_builtin(__builtin_amdgcn_sched_barrier)
#define SCHED_FENCE() __builtin_amdgcn_sched_barrier(0)
#else
#define SCHED_FENCE() asm volatile("" ::: "memory")
#endif

// Launder a wave-uniform pointer through an empty asm so LICM cannot hoist
// loads based on it out of a loop (keeps weights streaming from L2 instead of
// being materialized into a scratch spill array); rebuilt as a GLOBAL pointer.
__device__ __forceinline__ gc_h16* launder_global_s(const h16* p) {
  unsigned long long u = (unsigned long long)p;
  asm volatile("" : "+s"(u));
  return (gc_h16*)u;
}
// Same for a per-lane (divergent) pointer.
__device__ __forceinline__ gc_h16* launder_global_v(const h16* p) {
  unsigned long long u = (unsigned long long)p;
  asm volatile("" : "+v"(u));
  return (gc_h16*)u;
}
// Plain generic->global reinterpretation (no laundering).
__device__ __forceinline__ gc_h16* as_global_c(const h16* p) {
  return (gc_h16*)(unsigned long long)p;
}
__device__ __forceinline__ g_h16* as_global(h16* p) {
  return (g_h16*)(unsigned long long)p;
}

__device__ __forceinline__ float sig_(float x)  { return 1.0f / (1.0f + __expf(-x)); }
__device__ __forceinline__ float tanh_(float x) { float e = __expf(2.0f * x); return 1.0f - 2.0f / (e + 1.0f); }
__device__ __forceinline__ v8f wmma_f16(v16h a, v16h b, v8f c) {
  return __builtin_amdgcn_wmma_f32_16x16x32_f16(false, a, false, b, (short)0, c, false, false);
}

// ---------------------------------------------------------------- prep kernels

// xe[s][b][0:300] = emb[x[b][s]], padded with zeros to DP=320, f32 -> f16
__global__ void embed_kernel(const int* __restrict__ x, const float* __restrict__ emb,
                             h16* __restrict__ xe) {
  int i = blockIdx.x * 256 + threadIdx.x;          // < S*B*DP
  int d = i % DP_; int sb = i / DP_;
  int b = sb % B_; int s = sb / B_;
  float v = 0.0f;
  if (d < D_) v = emb[(size_t)x[b * S_ + s] * D_ + d];
  xe[i] = (h16)v;
}

// Wp[bd][n][k] : k<256 -> Whh[n][k], 256<=k<556 -> Wih[n][k-256], else 0  (f16)
__global__ void packw_kernel(const float* __restrict__ Wih_f, const float* __restrict__ Whh_f,
                             const float* __restrict__ Wih_b, const float* __restrict__ Whh_b,
                             h16* __restrict__ Wp) {
  int i = blockIdx.x * 256 + threadIdx.x;          // < NBD*G4*KP
  int k = i % KP_; int r = i / KP_;
  int n = r % G4_; int bd = r / G4_;
  int br = bd >> 1, dir = bd & 1;
  const float* Wih = dir ? Wih_b : Wih_f;
  const float* Whh = dir ? Whh_b : Whh_f;
  float v = 0.0f;
  if (k < H_)            v = Whh[((size_t)br * G4_ + n) * H_ + k];
  else if (k < H_ + D_)  v = Wih[((size_t)br * G4_ + n) * D_ + (k - H_)];
  Wp[i] = (h16)v;
}

__global__ void bias_kernel(const float* __restrict__ bih_f, const float* __restrict__ bhh_f,
                            const float* __restrict__ bih_b, const float* __restrict__ bhh_b,
                            float* __restrict__ bsum) {
  int i = blockIdx.x * 256 + threadIdx.x;          // < NBD*G4
  int n = i % G4_; int bd = i / G4_;
  int br = bd >> 1, dir = bd & 1;
  const float* bi = dir ? bih_b : bih_f;
  const float* bh = dir ? bhh_b : bhh_f;
  bsum[i] = bi[br * G4_ + n] + bh[br * G4_ + n];
}

__global__ void waconv_kernel(const float* __restrict__ Wa, h16* __restrict__ Wa_h) {
  int i = blockIdx.x * 256 + threadIdx.x;          // < NB*H2*H2
  Wa_h[i] = (h16)Wa[i];
}

// ------------------------------------------------- fused bidirectional LSTM scan
// One m-tile group: GEMM over fused K=[h(256)|x(320)] for all 4 gates of this
// wave's 16 h-columns, then the LSTM cell update entirely in registers.
// creg is a statically-indexed register array (separate array per peeled call).
__device__ __forceinline__ void lstm_group(
    const h16* __restrict__ hsrc,   // LDS: previous hidden state [B][H]
    gc_h16*    xrow,                // global: x_t [B][DP]
    gc_h16*    Wt,                  // global: packed weights [G4][KP] (laundered)
    float bg0, float bg1, float bg2, float bg3,
    float (&creg)[8],
    int mbase, int l, int hs, int w,
    h16* __restrict__ hdst,         // LDS: next hidden state [B][H]
    g_h16*     gdst)                // global: h_all row base (index by m*H2)
{
  const v8f vzero = {0.f,0.f,0.f,0.f,0.f,0.f,0.f,0.f};
  v8f acc[4] = {vzero, vzero, vzero, vzero};
  const int m = mbase + l;

#pragma unroll
  for (int kt = 0; kt < KP_ / 32; ++kt) {
    const int k0 = kt * 32;
    AF a;
    if (k0 < H_) {                       // A from LDS hidden state
      const h16* base = hsrc + m * H_ + k0 + hs * 8;
      a.v8[0] = *(const v8h*)(base);
      a.v8[1] = *(const v8h*)(base + 16);
    } else {                             // A from padded embedding (global)
      gc_h16* base = xrow + (size_t)m * DP_ + (k0 - H_) + hs * 8;
      a.v8[0] = *(gc_v8h*)(base);
      a.v8[1] = *(gc_v8h*)(base + 16);
    }
    v16h bf[4];                          // batch the 4 weight fragments: one
#pragma unroll                           // load clause, overlapped latencies
    for (int q = 0; q < 4; ++q) {
      const int n = q * H_ + 16 * w + l; // B-fragment: two 16B chunks per lane
      AF bt;
      bt.v8[0] = *(gc_v8h*)(Wt + (size_t)n * KP_ + k0 + hs * 8);
      bt.v8[1] = *(gc_v8h*)(Wt + (size_t)n * KP_ + k0 + 16 + hs * 8);
      bf[q] = bt.v16;
    }
#pragma unroll
    for (int q = 0; q < 4; ++q)
      acc[q] = wmma_f16(a.v16, bf[q], acc[q]);
  }

  const int hcol = 16 * w + l;
#pragma unroll
  for (int r = 0; r < 8; ++r) {
    const float gi = acc[0][r] + bg0;
    const float gf = acc[1][r] + bg1;
    const float gg = acc[2][r] + bg2;
    const float go = acc[3][r] + bg3;
    const float c  = sig_(gf) * creg[r] + sig_(gi) * tanh_(gg);
    creg[r] = c;
    const float h = sig_(go) * tanh_(c);
    const int m2 = mbase + r + 8 * hs;
    hdst[m2 * H_ + hcol] = (h16)h;
    gdst[(size_t)m2 * H2_] = (h16)h;
  }
}

// One workgroup per (branch, direction). 32 waves; wave (mh,w) owns m-tiles
// {2mh, 2mh+1} and h-columns [16w, 16w+16) across ALL FOUR gates. The two
// m-tiles are processed as two sequential passes (sched_barrier between) to
// keep live accumulators at 4 tiles (32 VGPRs) and avoid scratch spills.
__global__ __launch_bounds__(1024) void lstm_scan_kernel(
    const h16* __restrict__ xe,     // [S][B][DP]
    const h16* __restrict__ Wp,     // [NBD][G4][KP]
    const float* __restrict__ bsum, // [NBD][G4]
    h16* __restrict__ h_all)        // [NB][S][B][H2]
{
  const int bd  = blockIdx.x;              // 0..9
  const int br  = bd >> 1, dir = bd & 1;
  const int tid = threadIdx.x;
  const int wv  = tid >> 5;
  const int lane = tid & 31;
  const int l    = lane & 15;
  const int hs   = lane >> 4;
  const int w    = wv & 15;                // h-column tile
  const int mh   = wv >> 4;                // m-tile pair

  __shared__ h16 hbuf[2][B_][H_];          // ping-pong hidden state, 64 KB

  {
    h16* p = &hbuf[0][0][0];
    for (int i = tid; i < 2 * B_ * H_; i += blockDim.x) p[i] = (h16)0.0f;
  }
  __syncthreads();

  const h16* Wbd = Wp + (size_t)bd * G4_ * KP_;
  const float bg0 = bsum[bd * G4_ + 0 * H_ + 16 * w + l];
  const float bg1 = bsum[bd * G4_ + 1 * H_ + 16 * w + l];
  const float bg2 = bsum[bd * G4_ + 2 * H_ + 16 * w + l];
  const float bg3 = bsum[bd * G4_ + 3 * H_ + 16 * w + l];

  float creg0[8] = {};
  float creg1[8] = {};

  for (int t = 0; t < S_; ++t) {
    const int s = dir ? (S_ - 1 - t) : t;
    gc_h16* xrow = as_global_c(xe) + (size_t)s * B_ * DP_;
    const int p = t & 1;

    if (t + 1 < S_) {                       // prefetch next step's x tile (40KB)
      const int s2 = dir ? (S_ - 2 - t) : (t + 1);
      __builtin_prefetch(xe + (size_t)s2 * B_ * DP_ + tid * 32, 0, 1);
    }

    // Re-launder the weight pointer every step: forces the B-fragment loads to
    // stay inside the loop (L2-resident streaming, global_load path) instead
    // of being hoisted by LICM into a giant scratch-spill array.
    gc_h16* Wt = launder_global_s(Wbd);

    const h16* hsrc = &hbuf[p][0][0];
    h16*       hdst = &hbuf[1 - p][0][0];
    g_h16*     gdst = as_global(h_all) + ((size_t)br * S_ + s) * B_ * H2_
                      + dir * H_ + (16 * w + l);

    lstm_group(hsrc, xrow, Wt, bg0, bg1, bg2, bg3, creg0,
               (2 * mh + 0) * 16, l, hs, w, hdst, gdst);
    SCHED_FENCE();
    lstm_group(hsrc, xrow, Wt, bg0, bg1, bg2, bg3, creg1,
               (2 * mh + 1) * 16, l, hs, w, hdst, gdst);

    __syncthreads();                        // publish h for next step
  }
}

// ----------------------------------- attention score: s = tanh(h Wa^T + ba) . watt
// One wave per 16-row tile of flattened (s*B+b) rows; u never materialized.
// A-fragments: first K-half preloaded in registers, second half streamed from
// cache inside the jt loop (row pointer laundered per iteration so the stream
// is not LICM-hoisted back into registers/scratch).
__global__ __launch_bounds__(256) void attn_score_kernel(
    const h16* __restrict__ h_all, const h16* __restrict__ Wa_h,
    const float* __restrict__ ba, const float* __restrict__ watt,
    float* __restrict__ sc)
{
  const int br = blockIdx.y;
  const int wv = threadIdx.x >> 5, lane = threadIdx.x & 31;
  const int l = lane & 15, hs = lane >> 4;
  const int row0 = (blockIdx.x * 8 + wv) * 16;

  const h16* arow = h_all + (size_t)br * S_ * B_ * H2_ + (size_t)(row0 + l) * H2_;
  AF a0[8];                                 // preload K = 0..255
#pragma unroll
  for (int kt = 0; kt < 8; ++kt) {
    a0[kt].v8[0] = *(const v8h*)(arow + kt * 32 + hs * 8);
    a0[kt].v8[1] = *(const v8h*)(arow + kt * 32 + 16 + hs * 8);
  }
  const h16* Wb = Wa_h + (size_t)br * H2_ * H2_;
  const float* bav = ba + br * H2_;
  const float* wat = watt + br * H2_;
  const v8f vzero = {0.f,0.f,0.f,0.f,0.f,0.f,0.f,0.f};

  float rs[8] = {};
  for (int jt = 0; jt < 32; ++jt) {
    v8f acc = vzero;
    const int n = jt * 16 + l;
    const h16* brow = Wb + (size_t)n * H2_ + hs * 8;
#pragma unroll
    for (int kt = 0; kt < 8; ++kt)
      acc = wmma_f16(a0[kt].v16, __builtin_shufflevector(*(const v8h*)(brow + kt * 32), *(const v8h*)(brow + kt * 32 + 16), 0,1,2,3,4,5,6,7,8,9,10,11,12,13,14,15), acc);
    gc_h16* ar = launder_global_v(arow);    // keep second-half A loads in-loop
#pragma unroll
    for (int kt = 8; kt < 16; ++kt) {       // stream K = 256..511 (cache-hot)
      AF a;
      a.v8[0] = *(gc_v8h*)(ar + kt * 32 + hs * 8);
      a.v8[1] = *(gc_v8h*)(ar + kt * 32 + 16 + hs * 8);
      acc = wmma_f16(a.v16, __builtin_shufflevector(*(const v8h*)(brow + kt * 32), *(const v8h*)(brow + kt * 32 + 16), 0,1,2,3,4,5,6,7,8,9,10,11,12,13,14,15), acc);
    }
    const float bj = bav[n], wj = wat[n];
#pragma unroll
    for (int r = 0; r < 8; ++r) rs[r] += tanh_(acc[r] + bj) * wj;
  }
#pragma unroll
  for (int off = 1; off < 16; off <<= 1)
#pragma unroll
    for (int r = 0; r < 8; ++r) rs[r] += __shfl_xor(rs[r], off, 32);
  if (l == 0) {
#pragma unroll
    for (int r = 0; r < 8; ++r)
      sc[(size_t)br * S_ * B_ + row0 + r + 8 * hs] = rs[r];
  }
}

// softmax over S for each (branch, b); scores laid out [br][s*B+b], in place
__global__ __launch_bounds__(512) void softmax_kernel(float* __restrict__ sc) {
  const int b = blockIdx.x, br = blockIdx.y, s = threadIdx.x;
  float* p = sc + (size_t)br * S_ * B_;
  const float v = p[(size_t)s * B_ + b];
  __shared__ float red[S_];
  red[s] = v; __syncthreads();
  for (int off = 256; off > 0; off >>= 1) {
    if (s < off) red[s] = fmaxf(red[s], red[s + off]);
    __syncthreads();
  }
  const float mx = red[0]; __syncthreads();
  const float e = __expf(v - mx);
  red[s] = e; __syncthreads();
  for (int off = 256; off > 0; off >>= 1) {
    if (s < off) red[s] += red[s + off];
    __syncthreads();
  }
  p[(size_t)s * B_ + b] = e / red[0];
}

// out[b][d] = sum_s a[s] * h[s][b][d]; writes general (br==0) / specifics (br>=1)
__global__ __launch_bounds__(512) void attn_sum_kernel(
    const float* __restrict__ sc, const h16* __restrict__ h_all,
    float* __restrict__ out)
{
  const int b = blockIdx.x, br = blockIdx.y, d = threadIdx.x;
  __shared__ float av[S_];
  av[threadIdx.x] = sc[(size_t)br * S_ * B_ + (size_t)threadIdx.x * B_ + b];
  __syncthreads();
  gc_h16* hb = as_global_c(h_all) + (size_t)br * S_ * B_ * H2_ + (size_t)b * H2_ + d;
  float acc = 0.0f;
  for (int s = 0; s < S_; ++s) acc += av[s] * (float)hb[(size_t)s * B_ * H2_];
  float* dst = (br == 0) ? (out + 128 + (size_t)b * H2_ + d)
                         : (out + 128 + 32768 + (size_t)(br - 1) * 32768 + (size_t)b * H2_ + d);
  *dst = acc;
}

// final = concat([specifics[z[b]][b], general[b]]) @ Wfc^T + bfc
__global__ void final_kernel(const float* __restrict__ Wfc, const float* __restrict__ bfc,
                             const int* __restrict__ z, float* __restrict__ out) {
  const int t = threadIdx.x;
  if (t >= 128) return;
  const int b = t >> 1, col = t & 1;
  const float* gen  = out + 128 + (size_t)b * H2_;
  const float* spec = out + 128 + 32768 + (size_t)z[b] * 32768 + (size_t)b * H2_;
  const float* wrow = Wfc + col * G4_;
  float acc = bfc[col];
  for (int k = 0; k < H2_; ++k) acc += wrow[k] * spec[k];
  for (int k = 0; k < H2_; ++k) acc += wrow[H2_ + k] * gen[k];
  out[b * 2 + col] = acc;
}

// ---------------------------------------------------------------------- launch
extern "C" void kernel_launch(void* const* d_in, const int* in_sizes, int n_in,
                              void* d_out, int out_size, void* d_ws, size_t ws_size,
                              hipStream_t stream) {
  const int*   x     = (const int*)  d_in[0];
  const int*   z     = (const int*)  d_in[1];
  const float* emb   = (const float*)d_in[2];
  const float* Wih_f = (const float*)d_in[3];
  const float* Whh_f = (const float*)d_in[4];
  const float* bih_f = (const float*)d_in[5];
  const float* bhh_f = (const float*)d_in[6];
  const float* Wih_b = (const float*)d_in[7];
  const float* Whh_b = (const float*)d_in[8];
  const float* bih_b = (const float*)d_in[9];
  const float* bhh_b = (const float*)d_in[10];
  const float* Wa    = (const float*)d_in[11];
  const float* ba    = (const float*)d_in[12];
  const float* watt  = (const float*)d_in[13];
  const float* Wfc   = (const float*)d_in[14];
  const float* bfc   = (const float*)d_in[15];
  float* out = (float*)d_out;

  // workspace layout (256B-aligned, total ~204 MB; everything fully written
  // before being read — no dependence on poisoned contents)
  char*  ws    = (char*)d_ws;
  h16*   xe_h  = (h16*)  (ws + 0);           //  20,971,520 B  [S][B][DP] f16
  h16*   Wp    = (h16*)  (ws + 20971520);    //  11,796,480 B  [NBD][G4][KP] f16
  float* bsum  = (float*)(ws + 32768000);    //      40,960 B  [NBD][G4] f32
  h16*   Wa_h  = (h16*)  (ws + 32808960);    //   2,621,440 B  [NB][H2][H2] f16
  h16*   h_all = (h16*)  (ws + 35430400);    // 167,772,160 B  [NB][S][B][H2] f16
  float* sc    = (float*)(ws + 203202560);   //     655,360 B  [NB][S*B] f32

  embed_kernel <<<(S_ * B_ * DP_) / 256, 256, 0, stream>>>(x, emb, xe_h);
  packw_kernel <<<(NBD_ * G4_ * KP_) / 256, 256, 0, stream>>>(Wih_f, Whh_f, Wih_b, Whh_b, Wp);
  bias_kernel  <<<(NBD_ * G4_) / 256, 256, 0, stream>>>(bih_f, bhh_f, bih_b, bhh_b, bsum);
  waconv_kernel<<<(NB_ * H2_ * H2_) / 256, 256, 0, stream>>>(Wa, Wa_h);

  lstm_scan_kernel<<<NBD_, 1024, 0, stream>>>(xe_h, Wp, bsum, h_all);

  attn_score_kernel<<<dim3((S_ * B_) / 128, NB_), 256, 0, stream>>>(h_all, Wa_h, ba, watt, sc);
  softmax_kernel   <<<dim3(B_, NB_), S_, 0, stream>>>(sc);
  attn_sum_kernel  <<<dim3(B_, NB_), H2_, 0, stream>>>(sc, h_all, out);
  final_kernel     <<<1, 128, 0, stream>>>(Wfc, bfc, z, out);
}
